// EfficientCrossAttention_61194694034044
// MI455X (gfx1250) — matrix operand order repaired and checked by reading, not compiled, not run on hardware
//
#include <hip/hip_runtime.h>
#include <hip/hip_bf16.h>

// ---------------------------------------------------------------------------
// EfficientCrossAttention, MI455X (gfx1250), wave32 + WMMA bf16 + async LDS.
//
// B=8, C=256, H=W=128, MID=32, R=8  ->  N=16384, n=256 pooled positions.
// pool is linear => k = Wk*pool(y)+bk, v = Wv*pool(y)+bv  (huge reduction).
//
// Pipeline (all on `stream`):
//   K1 pool_kernel : y -> yp[b][j][256c]   (bf16, 8x8 block mean)
//   K2 wq_cvt      : Wq f32 -> bf16 row-major
//   K3 kv_kernel   : [Wk;Wv](288x256) x yp(256x256) per batch via WMMA bf16
//                    -> k2[b][j][m] bf16, v2[b][c][j] bf16 (+bias)
//   K4 attn_kernel : per (batch, 256-pos block): async-stage v2/k2/Wq in LDS,
//                    per wave: q = Wq*x (WMMA), s = q^T k (WMMA),
//                    softmax (shfl within 16-lane half), out = v*P^T (WMMA),
//                    epilogue out = gamma*out + x.
// ---------------------------------------------------------------------------

typedef __attribute__((ext_vector_type(16))) __bf16 v16bf;
typedef __attribute__((ext_vector_type(8)))  float  v8f;

#define B_    8
#define C_    256
#define HW_   16384
#define MID_  32
#define NP_   256     // pooled positions per batch (16*16)

__device__ inline v16bf mkvec(uint4 lo, uint4 hi) {
  union { uint4 q[2]; v16bf v; } u;
  u.q[0] = lo; u.q[1] = hi;
  return u.v;
}
__device__ inline uint4 ld16(const __bf16* p) { return *(const uint4*)p; }

__device__ inline v8f wmma_bf16(v16bf a, v16bf b, v8f c) {
  return __builtin_amdgcn_wmma_f32_16x16x32_bf16(false, a, false, b,
                                                 (short)0, c, false, false);
}

// Async DMA of 16 bytes global -> LDS (CDNA5 GLOBAL_LOAD_ASYNC_TO_LDS_B128).
// VDST = per-lane LDS byte offset, VADDR = per-lane 64-bit global address.
// Tracked by ASYNCcnt; dynamic LDS starts at offset 0 (no static LDS here),
// so the low 32 bits of the flat pointer are the in-allocation offset.
__device__ inline void async_cp16(__bf16* dst_lds, const __bf16* src_global) {
  unsigned off = (unsigned)(unsigned long long)(uintptr_t)dst_lds;
  asm volatile("global_load_async_to_lds_b128 %0, %1, off"
               :: "v"(off), "v"(src_global) : "memory");
}
__device__ inline void wait_async0() {
  asm volatile("s_wait_asynccnt 0" ::: "memory");
}
__device__ inline void wait_ds0() {
  asm volatile("s_wait_dscnt 0" ::: "memory");
}

// ---------------------------------------------------------------- K1: pool
// grid = B*C blocks, 256 threads. One (b,c) plane each; thread t = pooled pos.
// Writes yp[b][j][c] (transposed so K3's WMMA B-operand reads contiguous c).
__global__ void eca_pool(const float* __restrict__ y, __bf16* __restrict__ yp) {
  const int b = blockIdx.x >> 8;
  const int c = blockIdx.x & 255;
  const int t = threadIdx.x;            // j = ph*16 + pw
  const int ph = t >> 4, pw = t & 15;
  const float* base = y + (((size_t)(b * C_ + c)) << 14);
  float s = 0.f;
#pragma unroll
  for (int rr = 0; rr < 8; ++rr) {
    const float4* r4 = (const float4*)(base + (ph * 8 + rr) * 128 + pw * 8);
    float4 u = r4[0], v = r4[1];
    s += u.x + u.y + u.z + u.w + v.x + v.y + v.z + v.w;
  }
  s *= (1.f / 64.f);
  yp[((size_t)(b * NP_ + t)) * C_ + c] = (__bf16)s;
}

// ---------------------------------------------------------------- K2: Wq cvt
__global__ void eca_wqcvt(const float* __restrict__ Wq, __bf16* __restrict__ wq) {
  const int idx = blockIdx.x * 256 + threadIdx.x;
  if (idx < MID_ * C_) wq[idx] = (__bf16)Wq[idx];
}

// ---------------------------------------------------------------- K3: k2/v2
// grid = (18, B). m-tile mt covers rows [mt*16, mt*16+16) of [Wk;Wv] (288x256).
// 8 waves, each computes 2 j-tiles of the 16x16 D tile over K=256 (8 wmma ksteps).
__global__ void eca_kv(const __bf16* __restrict__ yp,
                       const float* __restrict__ Wk, const float* __restrict__ bk,
                       const float* __restrict__ Wv, const float* __restrict__ bv,
                       __bf16* __restrict__ k2, __bf16* __restrict__ v2) {
  __shared__ __bf16 w_s[16 * 264];                 // A-tile rows, padded
  const int mt = blockIdx.x, b = blockIdx.y;
  const int t = threadIdx.x;

  {  // stage 16 rows of [Wk;Wv] as bf16
    const int rr = t >> 4, col0 = (t & 15) * 16;
    const int row_g = mt * 16 + rr;
    const float* src = (row_g < MID_) ? (Wk + row_g * C_)
                                      : (Wv + (row_g - MID_) * C_);
#pragma unroll
    for (int i = 0; i < 16; ++i)
      w_s[rr * 264 + col0 + i] = (__bf16)src[col0 + i];
  }
  __syncthreads();

  const int wv = t >> 5, lane = t & 31;
  const int half = (lane >> 4) & 1, n16 = lane & 15;

  v8f acc0 = {0.f,0.f,0.f,0.f,0.f,0.f,0.f,0.f};
  v8f acc1 = acc0;
#pragma unroll
  for (int ks = 0; ks < 8; ++ks) {
    const __bf16* ap = w_s + n16 * 264 + ks * 32 + half * 8;
    v16bf a = mkvec(ld16(ap), ld16(ap + 16));
#pragma unroll
    for (int jj = 0; jj < 2; ++jj) {
      const int j = (wv * 2 + jj) * 16 + n16;
      const __bf16* bp = yp + (((size_t)(b * NP_ + j)) << 8) + ks * 32 + half * 8;
      v16bf bb = mkvec(ld16(bp), ld16(bp + 16));
      if (jj == 0) acc0 = wmma_bf16(a, bb, acc0);
      else         acc1 = wmma_bf16(a, bb, acc1);
    }
  }
#pragma unroll
  for (int jj = 0; jj < 2; ++jj) {
    const int j = (wv * 2 + jj) * 16 + n16;
#pragma unroll
    for (int r = 0; r < 8; ++r) {
      const int row_g = mt * 16 + r + 8 * half;
      const float val = ((jj == 0) ? acc0[r] : acc1[r]) +
                        ((row_g < MID_) ? bk[row_g] : bv[row_g - MID_]);
      if (row_g < MID_)
        k2[((size_t)(b * NP_ + j)) * MID_ + row_g] = (__bf16)val;        // [b][j][m]
      else
        v2[((size_t)(b * C_ + (row_g - MID_))) * NP_ + j] = (__bf16)val; // [b][c][j]
    }
  }
}

// ---------------------------------------------------------------- K4: fused
// grid = (64, B); 256 threads = 8 waves; each wave owns 2 p-tiles of 16 pos.
// Dynamic LDS: v_s[256][264] + k_s[256][40] + wq_s[32][264]
//              + per-wave st[16][264] (x-tile then P) + q_s[16][40].
__global__ void eca_attn(const float* __restrict__ x,
                         const __bf16* __restrict__ wq,
                         const float* __restrict__ bq,
                         const __bf16* __restrict__ k2,
                         const __bf16* __restrict__ v2,
                         const float* __restrict__ gamma,
                         float* __restrict__ out) {
  extern __shared__ char smem[];
  __bf16* v_s   = (__bf16*)smem;                  // 256*264
  __bf16* k_s   = v_s + 256 * 264;                // 256*40
  __bf16* wq_s  = k_s + 256 * 40;                 // 32*264
  __bf16* st_a  = wq_s + 32 * 264;                // 8 * 16*264 (x-tile, then P)
  __bf16* qs_a  = st_a + 8 * 16 * 264;            // 8 * 16*40

  const int b = blockIdx.y, pb = blockIdx.x, t = threadIdx.x;

  {  // async stage v2[b] (128KB bf16) into padded v_s
    const __bf16* src = v2 + (((size_t)b) << 16);
#pragma unroll
    for (int k = 0; k < 32; ++k) {
      const int idx = t + (k << 8);
      const int c = idx >> 5, seg = idx & 31;
      async_cp16(v_s + c * 264 + seg * 8, src + (c << 8) + seg * 8);
    }
  }
  {  // async stage k2[b] (16KB bf16) into padded k_s
    const __bf16* src = k2 + (((size_t)b) << 13);
#pragma unroll
    for (int k = 0; k < 4; ++k) {
      const int idx = t + (k << 8);
      const int j = idx >> 2, seg = idx & 3;
      async_cp16(k_s + j * 40 + seg * 8, src + (j << 5) + seg * 8);
    }
  }
  {  // async stage Wq bf16 (16KB) into padded wq_s
#pragma unroll
    for (int k = 0; k < 2; ++k) {
      const int idx = t + (k << 8);
      const int m = idx >> 4, seg = idx & 15;
      async_cp16(wq_s + m * 264 + seg * 8, wq + (m << 8) + seg * 8);
    }
  }
  wait_async0();
  __syncthreads();

  const int wv = t >> 5, lane = t & 31;
  const int half = (lane >> 4) & 1, n16 = lane & 15;
  __bf16* stw = st_a + wv * (16 * 264);
  __bf16* qsw = qs_a + wv * (16 * 40);
  const float g = gamma[0];

  for (int it = 0; it < 2; ++it) {
    const int p0 = pb * 256 + (wv * 2 + it) * 16;

    // ---- stage x tile as bf16, transposed st[p][c]; float4 loads -------
    const float* xb = x + (((size_t)b) << 22) + p0;
#pragma unroll 4
    for (int i = 0; i < 32; ++i) {
      const int idx = (i << 5) + lane;
      const int c = idx >> 2, pg = (idx & 3) << 2;
      const float4 v4 = *(const float4*)(xb + ((size_t)c << 14) + pg);
      stw[(pg + 0) * 264 + c] = (__bf16)v4.x;
      stw[(pg + 1) * 264 + c] = (__bf16)v4.y;
      stw[(pg + 2) * 264 + c] = (__bf16)v4.z;
      stw[(pg + 3) * 264 + c] = (__bf16)v4.w;
    }
    wait_ds0();                           // wave-local LDS ordering

    // ---- GEMM1: q[32][16] = Wq(32x256) * x_tile(256x16) ---------------
#pragma unroll
    for (int mt = 0; mt < 2; ++mt) {
      v8f acc = {0.f,0.f,0.f,0.f,0.f,0.f,0.f,0.f};
#pragma unroll
      for (int ks = 0; ks < 8; ++ks) {
        const __bf16* ap = wq_s + (mt * 16 + n16) * 264 + ks * 32 + half * 8;
        v16bf a = mkvec(ld16(ap), ld16(ap + 16));
        const __bf16* bp = stw + n16 * 264 + ks * 32 + half * 8;
        v16bf bb = mkvec(ld16(bp), ld16(bp + 16));
        acc = wmma_bf16(a, bb, acc);
      }
#pragma unroll
      for (int r = 0; r < 8; ++r) {       // D: M=r+8h (=m), N=n16 (=p)
        const int m = mt * 16 + r + 8 * half;
        qsw[n16 * 40 + m] = (__bf16)(acc[r] + bq[m]);   // q_s[p][m]
      }
    }
    wait_ds0();

    // ---- GEMM2: s[16p][256j] = q^T(16x32) * k(32x256) -----------------
    v16bf aq;
    {
      const __bf16* ap = qsw + n16 * 40 + half * 8;
      aq = mkvec(ld16(ap), ld16(ap + 16));
    }
    v8f sc[16];
#pragma unroll
    for (int jt = 0; jt < 16; ++jt) {
      const __bf16* bp = k_s + (jt * 16 + n16) * 40 + half * 8;
      v16bf bb = mkvec(ld16(bp), ld16(bp + 16));
      v8f z = {0.f,0.f,0.f,0.f,0.f,0.f,0.f,0.f};
      sc[jt] = wmma_bf16(aq, bb, z);      // lane holds s[r+8h][jt*16+n16]
    }

    // ---- softmax over j (row p = r+8h lives in one 16-lane half) ------
#pragma unroll
    for (int r = 0; r < 8; ++r) {
      float rm = sc[0][r];
#pragma unroll
      for (int jt = 1; jt < 16; ++jt) rm = fmaxf(rm, sc[jt][r]);
      rm = fmaxf(rm, __shfl_xor(rm, 1));
      rm = fmaxf(rm, __shfl_xor(rm, 2));
      rm = fmaxf(rm, __shfl_xor(rm, 4));
      rm = fmaxf(rm, __shfl_xor(rm, 8));
      float sum = 0.f;
#pragma unroll
      for (int jt = 0; jt < 16; ++jt) {
        const float e = __expf(sc[jt][r] - rm);
        sc[jt][r] = e;
        sum += e;
      }
      sum += __shfl_xor(sum, 1);
      sum += __shfl_xor(sum, 2);
      sum += __shfl_xor(sum, 4);
      sum += __shfl_xor(sum, 8);
      const float inv = 1.f / sum;
#pragma unroll
      for (int jt = 0; jt < 16; ++jt)     // P_s[p][j] (reuse stw)
        stw[(r + 8 * half) * 264 + jt * 16 + n16] = (__bf16)(sc[jt][r] * inv);
    }
    wait_ds0();

    // ---- prefetch next x tile into caches while GEMM3 runs ------------
    if (it == 0) {
      const float* xn = xb + 16;          // next p-tile of this wave
#pragma unroll
      for (int i = 0; i < 8; ++i)
        __builtin_prefetch(xn + (((size_t)(i * 32 + lane)) << 14), 0, 0);
    }

    // ---- GEMM3 + epilogue: out[c][p] = gamma * (v * P^T) + x ----------
    for (int ct = 0; ct < 16; ++ct) {
      v8f acc = {0.f,0.f,0.f,0.f,0.f,0.f,0.f,0.f};
#pragma unroll
      for (int ks = 0; ks < 8; ++ks) {
        const __bf16* ap = v_s + (ct * 16 + n16) * 264 + ks * 32 + half * 8;
        v16bf a = mkvec(ld16(ap), ld16(ap + 16));
        const __bf16* bp = stw + n16 * 264 + ks * 32 + half * 8;
        v16bf bb = mkvec(ld16(bp), ld16(bp + 16));
        acc = wmma_bf16(a, bb, acc);
      }
#pragma unroll
      for (int r = 0; r < 8; ++r) {       // D: M=c_local, N=p
        const int c = ct * 16 + r + 8 * half;
        const size_t off = (((size_t)(b * C_ + c)) << 14) + p0 + n16;
        out[off] = g * acc[r] + x[off];
      }
    }
    wait_ds0();                           // before stw reuse
  }
}

// ---------------------------------------------------------------------------
extern "C" void kernel_launch(void* const* d_in, const int* in_sizes, int n_in,
                              void* d_out, int out_size, void* d_ws, size_t ws_size,
                              hipStream_t stream) {
  const float* x     = (const float*)d_in[0];
  const float* y     = (const float*)d_in[1];
  const float* Wq    = (const float*)d_in[2];
  const float* bq    = (const float*)d_in[3];
  const float* Wk    = (const float*)d_in[4];
  const float* bk    = (const float*)d_in[5];
  const float* Wv    = (const float*)d_in[6];
  const float* bv    = (const float*)d_in[7];
  const float* gamma = (const float*)d_in[8];
  float* out = (float*)d_out;

  // workspace carve (bf16): yp[B][256][256], wq[32][256], k2[B][256][32], v2[B][256][256]
  __bf16* yp  = (__bf16*)d_ws;
  __bf16* wqb = yp  + (size_t)B_ * NP_ * C_;
  __bf16* k2  = wqb + (size_t)MID_ * C_;
  __bf16* v2  = k2  + (size_t)B_ * NP_ * MID_;

  eca_pool <<<dim3(B_ * C_), 256, 0, stream>>>(y, yp);
  eca_wqcvt<<<dim3(32),      256, 0, stream>>>(Wq, wqb);
  eca_kv   <<<dim3(18, B_),  256, 0, stream>>>(yp, Wk, bk, Wv, bv, k2, v2);

  const size_t smem = (size_t)(256 * 264 + 256 * 40 + 32 * 264 +
                               8 * 16 * 264 + 8 * 16 * 40) * sizeof(__bf16);
  (void)hipFuncSetAttribute((const void*)eca_attn,
                            hipFuncAttributeMaxDynamicSharedMemorySize, (int)smem);
  eca_attn <<<dim3(64, B_),  256, smem, stream>>>(x, wqb, bq, k2, v2, gamma, out);
}
